// LSHAttention_60644938219942
// MI455X (gfx1250) — matrix operand order repaired and checked by reading, not compiled, not run on hardware
//
#include <hip/hip_runtime.h>

#define B_ 16
#define S_ 4096
#define D_ 64
#define H_ 8
#define NB_ 64                 // buckets per hash
#define NBINS_ (H_ * NB_)      // 512 buckets per batch
#define TOT_ (H_ * S_)         // 32768 items per batch
#define NCHUNK_ 512            // chunks per batch
#define BS_ 64                 // rows per chunk
#define KW_ 128                // keys per chunk (look-one-back)

typedef __attribute__((ext_vector_type(16))) __bf16 bfrag16;
typedef __attribute__((ext_vector_type(8)))  float  f32x8;
typedef __attribute__((ext_vector_type(4)))  unsigned int u32x4;

union FragU { bfrag16 v; u32x4 u[2]; };

static __device__ __forceinline__ unsigned short f2bf(float x) {
  unsigned int u = __float_as_uint(x);
  unsigned int r = (u + 0x7FFFu + ((u >> 16) & 1u)) >> 16;   // RNE
  return (unsigned short)r;
}

// A-matrix 16x32 bf16 fragment (doc layout): lane l -> M = m*16 + (l&15);
// K base = kc*32 + (l&16 ? 8 : 0); elements 0..7 at K..K+7, 8..15 at K+16..K+23.
static __device__ __forceinline__ bfrag16
load_fragA(const unsigned short* base, int stride, int m, int kc, int lane) {
  const unsigned short* p =
      base + (m * 16 + (lane & 15)) * stride + kc * 32 + ((lane & 16) ? 8 : 0);
  FragU f;
  f.u[0] = *reinterpret_cast<const u32x4*>(p);
  f.u[1] = *reinterpret_cast<const u32x4*>(p + 16);
  return f.v;
}

// B-matrix 32x16 bf16 fragment: lane l -> N = n*16 + (l&15);
// K base = kc*32 + (l&16 ? 8 : 0); elements 0..7 at K..K+7, 8..15 at K+16..K+23 (rows of LDS = N index).
static __device__ __forceinline__ bfrag16
load_fragB(const unsigned short* base, int stride, int n, int kc, int lane) {
  const unsigned short* p =
      base + (n * 16 + (lane & 15)) * stride + kc * 32 + ((lane & 16) ? 8 : 0);
  FragU f;
  f.u[0] = *reinterpret_cast<const u32x4*>(p);
  f.u[1] = *reinterpret_cast<const u32x4*>(p + 16);
  return f.v;
}

static __device__ __forceinline__ void atomic_add_f32(float* p, float v) {
  __hip_atomic_fetch_add(p, v, __ATOMIC_RELAXED, __HIP_MEMORY_SCOPE_AGENT);
}

// ---------------- Kernel 1: LSH hashing ----------------
// grid (B*H, S/256), block 256. rotations: [D][H][32]
__global__ void lsh_hash_kernel(const float* __restrict__ qk,
                                const float* __restrict__ rot,
                                int* __restrict__ buckets) {
  __shared__ float rotL[D_ * 32];
  const int bh = blockIdx.x;
  const int b = bh >> 3, h = bh & 7;
  const int tid = threadIdx.x;
  for (int idx = tid; idx < D_ * 32; idx += 256) {
    int f = idx >> 5, i = idx & 31;
    rotL[idx] = rot[f * (H_ * 32) + h * 32 + i];
  }
  __syncthreads();
  const int t = blockIdx.y * 256 + tid;
  const float* q = qk + ((size_t)b * S_ + t) * D_;
  float acc[32];
#pragma unroll
  for (int i = 0; i < 32; ++i) acc[i] = 0.f;
  for (int f = 0; f < D_; ++f) {
    float qf = q[f];
    const float* rr = &rotL[f * 32];
#pragma unroll
    for (int i = 0; i < 32; ++i) acc[i] += qf * rr[i];
  }
  float bv = acc[0];
  int bi = 0;
#pragma unroll
  for (int i = 1; i < 32; ++i)
    if (acc[i] > bv) { bv = acc[i]; bi = i; }
#pragma unroll
  for (int i = 0; i < 32; ++i)
    if (-acc[i] > bv) { bv = -acc[i]; bi = 32 + i; }
  buckets[(size_t)b * TOT_ + h * S_ + t] = bi + h * NB_;
}

// ---------------- Kernel 2: per-batch histogram + exclusive prefix ----------------
// grid B, block 512
__global__ void lsh_hist_kernel(const int* __restrict__ buckets,
                                int* __restrict__ offsets) {
  __shared__ int cnt[NBINS_];
  const int b = blockIdx.x, tid = threadIdx.x;
  cnt[tid] = 0;
  __syncthreads();
  for (int idx = tid; idx < TOT_; idx += 512)
    atomicAdd(&cnt[buckets[(size_t)b * TOT_ + idx]], 1);
  __syncthreads();
  if (tid == 0) {
    int run = 0;
    for (int k = 0; k < NBINS_; ++k) { int c = cnt[k]; cnt[k] = run; run += c; }
  }
  __syncthreads();
  offsets[b * NBINS_ + tid] = cnt[tid];
}

// ---------------- Kernel 3: stable rank-scatter (counting sort) ----------------
// grid B*H, block 128; each thread owns 32 consecutive t.
__global__ void lsh_rank_kernel(const int* __restrict__ buckets,
                                const int* __restrict__ offsets,
                                int* __restrict__ sorted_i) {
  __shared__ int lc[128 * NB_];   // 32 KB
  const int bh = blockIdx.x;
  const int b = bh >> 3, h = bh & 7;
  const int tid = threadIdx.x;
  const int t0 = tid * 32;
  for (int k = 0; k < NB_; ++k) lc[tid * NB_ + k] = 0;
  const int* bk = buckets + (size_t)b * TOT_ + h * S_;
  for (int dt = 0; dt < 32; ++dt) lc[tid * NB_ + (bk[t0 + dt] - h * NB_)]++;
  __syncthreads();
  if (tid < NB_) {
    int run = 0;
    for (int th = 0; th < 128; ++th) {
      int c = lc[th * NB_ + tid];
      lc[th * NB_ + tid] = run;
      run += c;
    }
  }
  __syncthreads();
  for (int dt = 0; dt < 32; ++dt) {
    int t = t0 + dt;
    int g = bk[t];
    int lb = g - h * NB_;
    int r = lc[tid * NB_ + lb];
    lc[tid * NB_ + lb] = r + 1;
    int pos = offsets[b * NBINS_ + g] + r;
    sorted_i[(size_t)b * TOT_ + pos] = h * S_ + t;
  }
}

// ---------------- shared staging for attention ----------------
// stages Q (scaled by D^-1/2, bf16), normalized K rows (bf16), tq/tk, row ids.
static __device__ __forceinline__ void stage_qk(
    const float* __restrict__ qk, const int* __restrict__ sorted_i,
    int b, int c, int tid,
    unsigned short* Qs, unsigned short* Ks,
    int* rows_i, int* tq, int* tk) {
  const int cp = (c + NCHUNK_ - 1) & (NCHUNK_ - 1);
  if (tid < 64) {
    int i = sorted_i[(size_t)b * TOT_ + c * BS_ + tid];
    rows_i[tid] = i;
    int t = i & (S_ - 1);
    tq[tid] = t;
    tk[tid] = t;
  } else if (tid < 128) {
    int i2 = sorted_i[(size_t)b * TOT_ + cp * BS_ + (tid - 64)];
    tk[tid] = i2 & (S_ - 1);
  }
  __syncthreads();
  // Q: 64x64, each thread 16 elems, scale by 0.125 = D^-0.5
  {
    int r = tid >> 2, sc = (tid & 3) * 16;
    const float* src = qk + ((size_t)b * S_ + tq[r]) * D_ + sc;
#pragma unroll
    for (int k = 0; k < 16; ++k) Qs[r * D_ + sc + k] = f2bf(src[k] * 0.125f);
  }
  // K: 128 rows, L2-normalized
  if (tid < 128) {
    const float* src = qk + ((size_t)b * S_ + tk[tid]) * D_;
    float ss = 0.f;
#pragma unroll
    for (int f = 0; f < D_; ++f) { float x = src[f]; ss += x * x; }
    float rn = 1.f / fmaxf(sqrtf(ss), 1e-12f);
#pragma unroll
    for (int f = 0; f < D_; ++f) Ks[tid * D_ + f] = f2bf(src[f] * rn);
  }
}

// ---------------- Kernel 4: attention pass A -> LSE per sorted row ----------------
// grid B*512, block 256 (8 waves)
__global__ void lsh_attn_lse_kernel(const float* __restrict__ qk,
                                    const int* __restrict__ sorted_i,
                                    float* __restrict__ logits) {
  __shared__ alignas(16) unsigned short Qs[BS_ * D_];    // 8 KB
  __shared__ alignas(16) unsigned short Ks[KW_ * D_];    // 16 KB
  __shared__ float dotsS[BS_ * KW_];                     // 32 KB
  __shared__ int rows_i[BS_];
  __shared__ int tq[BS_];
  __shared__ int tk[KW_];

  const int bx = blockIdx.x;
  const int b = bx >> 9, c = bx & (NCHUNK_ - 1);
  const int tid = threadIdx.x;
  stage_qk(qk, sorted_i, b, c, tid, Qs, Ks, rows_i, tq, tk);
  __syncthreads();

  const int wv = tid >> 5, lane = tid & 31;
#pragma unroll
  for (int q = 0; q < 4; ++q) {
    int tile = wv * 4 + q;
    int m = tile >> 3, n = tile & 7;
    f32x8 acc = {0.f, 0.f, 0.f, 0.f, 0.f, 0.f, 0.f, 0.f};
#pragma unroll
    for (int kc = 0; kc < 2; ++kc) {
      bfrag16 a = load_fragA(Qs, D_, m, kc, lane);
      bfrag16 bb = load_fragB(Ks, D_, n, kc, lane);
      acc = __builtin_amdgcn_wmma_f32_16x16x32_bf16(
          false, a, false, bb, (short)0, acc, false, false);
    }
    int col = n * 16 + (lane & 15);
    int rbase = m * 16 + ((lane & 16) ? 8 : 0);
#pragma unroll
    for (int r = 0; r < 8; ++r) {
      int row = rbase + r;
      float d = acc[r];
      if (tq[row] == tk[col]) d = -1e5f;
      dotsS[row * KW_ + col] = d;
    }
  }
  __syncthreads();
  if (tid < BS_) {
    const float* dr = &dotsS[tid * KW_];
    float mx = -3.4e38f;
    for (int j = 0; j < KW_; ++j) mx = fmaxf(mx, dr[j]);
    float s = 0.f;
    for (int j = 0; j < KW_; ++j) s += __expf(dr[j] - mx);
    logits[(size_t)b * TOT_ + rows_i[tid]] = mx + __logf(s);
  }
}

// ---------------- Kernel 5: combination weights over hash rounds ----------------
// grid B*S/256, block 256
__global__ void lsh_weight_kernel(const float* __restrict__ logits,
                                  float* __restrict__ wts) {
  const int id = blockIdx.x * 256 + threadIdx.x;   // (b, t)
  const int b = id >> 12, t = id & (S_ - 1);
  float l[H_];
  float mx = -3.4e38f;
#pragma unroll
  for (int h = 0; h < H_; ++h) {
    l[h] = logits[(size_t)b * TOT_ + h * S_ + t];
    mx = fmaxf(mx, l[h]);
  }
  float s = 0.f;
#pragma unroll
  for (int h = 0; h < H_; ++h) s += __expf(l[h] - mx);
  float inv = 1.f / s;
#pragma unroll
  for (int h = 0; h < H_; ++h)
    wts[(size_t)b * TOT_ + h * S_ + t] = __expf(l[h] - mx) * inv;
}

// ---------------- Kernel 6: attention pass B -> weighted output accumulate ----------------
// grid B*512, block 256
__global__ void lsh_attn_out_kernel(const float* __restrict__ qk,
                                    const float* __restrict__ v,
                                    const int* __restrict__ sorted_i,
                                    const float* __restrict__ logits,
                                    const float* __restrict__ wts,
                                    float* __restrict__ out) {
  __shared__ alignas(16) unsigned short Qs[BS_ * D_];      // 8 KB
  __shared__ alignas(16) unsigned short Ks[KW_ * D_];      // 16 KB
  __shared__ alignas(16) unsigned short VsT[D_ * KW_];     // 16 KB (V transposed)
  __shared__ alignas(16) unsigned short probsS[BS_ * KW_]; // 16 KB
  __shared__ int rows_i[BS_];
  __shared__ int tq[BS_];
  __shared__ int tk[KW_];
  __shared__ float lseS[BS_];
  __shared__ float wS[BS_];

  const int bx = blockIdx.x;
  const int b = bx >> 9, c = bx & (NCHUNK_ - 1);
  const int tid = threadIdx.x;
  stage_qk(qk, sorted_i, b, c, tid, Qs, Ks, rows_i, tq, tk);
  if (tid < 64) {
    int i = rows_i[tid];
    lseS[tid] = logits[(size_t)b * TOT_ + i];
    wS[tid] = wts[(size_t)b * TOT_ + i];
  }
  // V^T staging: threads 128..255 each own one key row jv, scatter to columns
  if (tid >= 128) {
    int jv = tid - 128;
    const float* src = v + ((size_t)b * S_ + tk[jv]) * D_;
#pragma unroll
    for (int f = 0; f < D_; ++f) VsT[f * KW_ + jv] = f2bf(src[f]);
  }
  __syncthreads();

  const int wv = tid >> 5, lane = tid & 31;
  // QK^T -> probs (weight folded in)
#pragma unroll
  for (int q = 0; q < 4; ++q) {
    int tile = wv * 4 + q;
    int m = tile >> 3, n = tile & 7;
    f32x8 acc = {0.f, 0.f, 0.f, 0.f, 0.f, 0.f, 0.f, 0.f};
#pragma unroll
    for (int kc = 0; kc < 2; ++kc) {
      bfrag16 a = load_fragA(Qs, D_, m, kc, lane);
      bfrag16 bb = load_fragB(Ks, D_, n, kc, lane);
      acc = __builtin_amdgcn_wmma_f32_16x16x32_bf16(
          false, a, false, bb, (short)0, acc, false, false);
    }
    int col = n * 16 + (lane & 15);
    int rbase = m * 16 + ((lane & 16) ? 8 : 0);
#pragma unroll
    for (int r = 0; r < 8; ++r) {
      int row = rbase + r;
      float p = (tq[row] == tk[col])
                    ? 0.f
                    : __expf(acc[r] - lseS[row]) * wS[row];
      probsS[row * KW_ + col] = f2bf(p);
    }
  }
  __syncthreads();

  // P @ V : 4x4 output tiles over 64x64, k = 128 in 4 chunks; 2 tiles per wave
#pragma unroll
  for (int q = 0; q < 2; ++q) {
    int tile = wv * 2 + q;
    int m = tile >> 2, n = tile & 3;
    f32x8 acc = {0.f, 0.f, 0.f, 0.f, 0.f, 0.f, 0.f, 0.f};
#pragma unroll
    for (int kc = 0; kc < 4; ++kc) {
      bfrag16 a = load_fragA(probsS, KW_, m, kc, lane);
      bfrag16 bb = load_fragB(VsT, KW_, n, kc, lane);
      acc = __builtin_amdgcn_wmma_f32_16x16x32_bf16(
          false, a, false, bb, (short)0, acc, false, false);
    }
    int col = n * 16 + (lane & 15);
    int rbase = m * 16 + ((lane & 16) ? 8 : 0);
#pragma unroll
    for (int r = 0; r < 8; ++r) {
      int row = rbase + r;
      int t = rows_i[row] & (S_ - 1);
      atomic_add_f32(&out[((size_t)b * S_ + t) * D_ + col], acc[r]);
    }
  }
}

extern "C" void kernel_launch(void* const* d_in, const int* in_sizes, int n_in,
                              void* d_out, int out_size, void* d_ws, size_t ws_size,
                              hipStream_t stream) {
  const float* qk  = (const float*)d_in[0];
  const float* v   = (const float*)d_in[1];
  const float* rot = (const float*)d_in[2];

  float* out = (float*)d_out;                                   // [B,S,D] f32
  int* buckets_out = (int*)((char*)d_out +
                            (size_t)B_ * S_ * D_ * sizeof(float)); // [B,H*S] i32

  // workspace carve-up (~6 MB)
  int* offsets   = (int*)d_ws;                                  // B*512
  int* sorted_i  = offsets + B_ * NBINS_;                       // B*32768
  float* logits  = (float*)(sorted_i + (size_t)B_ * TOT_);      // B*32768
  float* wts     = logits + (size_t)B_ * TOT_;                  // B*32768

  hipMemsetAsync(out, 0, (size_t)B_ * S_ * D_ * sizeof(float), stream);

  lsh_hash_kernel<<<dim3(B_ * H_, S_ / 256), 256, 0, stream>>>(qk, rot, buckets_out);
  lsh_hist_kernel<<<B_, NBINS_, 0, stream>>>(buckets_out, offsets);
  lsh_rank_kernel<<<B_ * H_, 128, 0, stream>>>(buckets_out, offsets, sorted_i);
  lsh_attn_lse_kernel<<<B_ * NCHUNK_, 256, 0, stream>>>(qk, sorted_i, logits);
  lsh_weight_kernel<<<B_ * S_ / 256, 256, 0, stream>>>(logits, wts);
  lsh_attn_out_kernel<<<B_ * NCHUNK_, 256, 0, stream>>>(qk, v, sorted_i, logits,
                                                        wts, out);
}
